// ComplexSSMState_71657234366912
// MI455X (gfx1250) — matrix operand order repaired and checked by reading, not compiled, not run on hardware
//
#include <hip/hip_runtime.h>
#include <hip/hip_bf16.h>
#include <cstdint>
#include <cstddef>

typedef __bf16 bf16;
typedef __bf16 v16bf __attribute__((ext_vector_type(16)));
typedef float  v8f   __attribute__((ext_vector_type(8)));
typedef unsigned int v4u __attribute__((ext_vector_type(4)));
typedef unsigned int v8u __attribute__((ext_vector_type(8)));

static constexpr int Bb = 4, Ss = 4096, Gg = 4, Dd = 256, Nn = 128;
static constexpr int M  = Bb * Ss * Gg;   // 65536 tokens
static constexpr int K1 = 2 * Dd;         // 512  (xcat width)
static constexpr int N1 = 2 * Nn;         // 256  (Bxr|Bxi)
static constexpr int K2 = 2 * Nn;         // 256  (hr|hi)
static constexpr int N2 = 2 * Dd;         // 512  (yr|yi)
static constexpr int NC = 32;             // scan chunks
static constexpr int LC = Ss / NC;        // 128 steps per chunk
static constexpr int NSCAN = Bb * Gg * Nn; // 2048 recurrences

union Frag16 {            // 16 bf16 = 32 bytes, moved as 2x uint4
    uint4 u[2];
    v16bf v;
    bf16  h[16];
};

__device__ __forceinline__ float clip_dt(float v) {
    return fminf(fmaxf(v, 1.0e-4f), 2.0f);
}

// --- CDNA5 async copy: global -> LDS without touching VGPRs (ASYNCcnt) ----
// lds_off is the byte offset inside the kernel's (single) LDS allocation.
__device__ __forceinline__ void async_ld_b128(unsigned lds_off, const void* g) {
    unsigned long long ga = (unsigned long long)(uintptr_t)g;
    asm volatile("global_load_async_to_lds_b128 %0, %1, off"
                 :: "v"(lds_off), "v"(ga) : "memory");
}
__device__ __forceinline__ void wait_async0() {
    asm volatile("s_wait_asynccnt 0x0" ::: "memory");
}

// --- CDNA5 Tensor Data Mover: one descriptor moves a whole 2D tile --------
// global (rowDw dwords per row, nRows rows, contiguous) -> LDS at lds_off,
// with native LDS row padding: after every (1<<(padIntv+1)) dwords stored,
// insert (padAmt+1) dwords of pad.  Tracked by TENSORcnt.
__device__ __forceinline__ void tdm_load_2d(unsigned lds_off, const void* g,
                                            unsigned rowDw, unsigned nRows,
                                            unsigned padIntv, unsigned padAmt) {
    unsigned long long ga = (unsigned long long)(uintptr_t)g;
    v4u g0;
    g0[0] = 1u;                                   // count=1, user descriptor
    g0[1] = lds_off;                              // lds_addr (bytes)
    g0[2] = (unsigned)ga;                         // global_addr[31:0]
    g0[3] = (unsigned)(ga >> 32) | (2u << 30);    // global_addr[56:32] | type=2
    v8u g1;
    g1[0] = (2u << 16)                            // data_size = 4B
          | (1u << 20)                            // pad_enable
          | (padIntv << 22) | (padAmt << 25);
    g1[1] = (rowDw & 0xFFFFu) << 16;              // tensor_dim0[15:0]
    g1[2] = (rowDw >> 16) | ((nRows & 0xFFFFu) << 16);  // dim0 hi | dim1 lo
    g1[3] = (nRows >> 16) | (rowDw << 16);        // dim1 hi | tile_dim0
    g1[4] = nRows & 0xFFFFu;                      // tile_dim1 (tile_dim2 = 0)
    g1[5] = rowDw;                                // tensor_dim0_stride[31:0]
    g1[6] = 0u;                                   // stride0 hi | stride1 lo
    g1[7] = 0u;                                   // stride1 hi
    asm volatile("tensor_load_to_lds %0, %1, null, null"
                 :: "s"(g0), "s"(g1) : "memory");
}
__device__ __forceinline__ void wait_tensor0() {
    __builtin_amdgcn_s_wait_tensorcnt(0);
}

// ---------------------------------------------------------------------------
// Kernel 0: pack complex weights into bf16 WMMA B-operand fragments.
// Fragment order (per 32x16 K-N tile): lane L holds column n = L&15,
// K-values k = kt*32 + (L>>4)*16 + e for e=0..15, stored contiguously.
// W_B[512,256]: [ [Br^T  Bi^T] ; [-Bi^T  Br^T] ]  (x=[xr|xi] -> [Bxr|Bxi])
// W_C[256,512]: [ [Cr    Ci  ] ; [-Ci    Cr  ] ]  (h=[hr|hi] -> [yr|yi])
// ---------------------------------------------------------------------------
__global__ void prep_weights(const float* __restrict__ Bwr, const float* __restrict__ Bwi,
                             const float* __restrict__ Cwr, const float* __restrict__ Cwi,
                             bf16* __restrict__ wbf, bf16* __restrict__ wcf) {
    const int tile = blockIdx.x;
    const int L = threadIdx.x;               // 0..31
    Frag16 f;
    if (tile < 256) {                        // W_B: 16 k-tiles x 16 n-tiles
        const int kt = tile >> 4, nt = tile & 15;
        const int nn = nt * 16 + (L & 15);
        const int kbase = kt * 32 + (L >> 4) * 16;
#pragma unroll
        for (int e = 0; e < 16; ++e) {
            const int k = kbase + e;
            float v;
            if (k < Dd) { const int d = k;
                v = (nn < Nn) ?  Bwr[nn * Dd + d] : Bwi[(nn - Nn) * Dd + d];
            } else {      const int d = k - Dd;
                v = (nn < Nn) ? -Bwi[nn * Dd + d] : Bwr[(nn - Nn) * Dd + d];
            }
            f.h[e] = (bf16)v;
        }
        uint4* dst = (uint4*)(wbf + ((size_t)tile * 32 + L) * 16);
        dst[0] = f.u[0]; dst[1] = f.u[1];
    } else {                                 // W_C: 8 k-tiles x 32 n-tiles
        const int t = tile - 256;
        const int kt = t >> 5, nt = t & 31;
        const int mm = nt * 16 + (L & 15);
        const int kbase = kt * 32 + (L >> 4) * 16;
#pragma unroll
        for (int e = 0; e < 16; ++e) {
            const int k = kbase + e;
            float v;
            if (k < Nn) { const int n = k;
                v = (mm < Dd) ?  Cwr[mm * Nn + n] : Cwi[(mm - Dd) * Nn + n];
            } else {      const int n = k - Nn;
                v = (mm < Dd) ? -Cwi[mm * Nn + n] : Cwr[(mm - Dd) * Nn + n];
            }
            f.h[e] = (bf16)v;
        }
        uint4* dst = (uint4*)(wcf + ((size_t)t * 32 + L) * 16);
        dst[0] = f.u[0]; dst[1] = f.u[1];
    }
}

// ---------------------------------------------------------------------------
// Kernel 1: one wave per token. Convert [xr|xi] -> bf16 xcat[M,512] and
// compute dt_mag/dt_phase = clip(exp(dt_w @ xcat + dt_b)) via wave reduction.
// ---------------------------------------------------------------------------
__global__ void prep_x_dt(const float* __restrict__ xr, const float* __restrict__ xi,
                          const float* __restrict__ dtw, const float* __restrict__ dtb,
                          bf16* __restrict__ xcat, float* __restrict__ dtm,
                          float* __restrict__ dtp) {
    const int m = (blockIdx.x << 3) + (threadIdx.x >> 5);   // token id
    const int L = threadIdx.x & 31;
    const float* src = (L < 16) ? (xr + (size_t)m * Dd + L * 16)
                                : (xi + (size_t)m * Dd + (L - 16) * 16);
    const int kidx = L * 16;                                // 0..511 in xcat
    const float4* p = (const float4*)src;
    float4 a0 = p[0], a1 = p[1], a2 = p[2], a3 = p[3];
    float xs[16] = {a0.x, a0.y, a0.z, a0.w, a1.x, a1.y, a1.z, a1.w,
                    a2.x, a2.y, a2.z, a2.w, a3.x, a3.y, a3.z, a3.w};
    float acc0 = 0.f, acc1 = 0.f;
    const float* w0 = dtw + kidx;
    const float* w1 = dtw + 512 + kidx;
    Frag16 f;
#pragma unroll
    for (int j = 0; j < 16; ++j) {
        acc0 += xs[j] * w0[j];
        acc1 += xs[j] * w1[j];
        f.h[j] = (bf16)xs[j];
    }
    uint4* dst = (uint4*)(xcat + (size_t)m * K1 + kidx);
    dst[0] = f.u[0]; dst[1] = f.u[1];
#pragma unroll
    for (int o = 16; o >= 1; o >>= 1) {
        acc0 += __shfl_xor(acc0, o, 32);
        acc1 += __shfl_xor(acc1, o, 32);
    }
    if (L == 0) {
        dtm[m] = clip_dt(__expf(acc0 + dtb[0]));
        dtp[m] = clip_dt(__expf(acc1 + dtb[1]));
    }
}

// ---------------------------------------------------------------------------
// Generic bf16 WMMA GEMM: Out[M,NTOT] = A[M,KDIM] (bf16, row major, staged
// into LDS by TDM or async-to-LDS) x Wfrag (pre-swizzled bf16 fragments),
// fp32 accum via v_wmma_f32_16x16x32_bf16. Each wave owns a 16(M) x 128(N)
// strip (8 tiles). SCALE: multiply each output row by dtm[row].
// ---------------------------------------------------------------------------
template <int KDIM, int NTOT, int MBLK, int WM, int WN, bool SCALE, bool TDM>
__global__ __launch_bounds__(WM * WN * 32)
void gemm_bf16_wmma(const bf16* __restrict__ Amat, const bf16* __restrict__ Wfrag,
                    const float* __restrict__ dtm, float* __restrict__ Out) {
    constexpr int KT = KDIM / 32;           // k-steps
    constexpr int NT = NTOT / 16;           // n-tiles total
    constexpr int LDSK = KDIM + 8;          // bf16 row stride (pad vs bank conflicts)
    constexpr int NTHREADS = WM * WN * 32;
    __shared__ bf16 lds[MBLK * LDSK];       // single LDS object -> base offset 0

    const int tid = threadIdx.x;
    const int m0 = blockIdx.x * MBLK;

    if (TDM) {
        // One TDM descriptor copies the whole MBLK x KDIM tile, inserting the
        // 4-dword LDS pad after every row (rowDw dwords) natively.
        constexpr unsigned rowDw = KDIM / 2;                   // 256 or 128
        constexpr unsigned padIntv = (rowDw == 256) ? 7u : 6u; // 256/128 dwords
        if (tid == 0) {
            tdm_load_2d(0u, Amat + (size_t)m0 * KDIM, rowDw, MBLK, padIntv, 3u);
            wait_tensor0();                 // s_wait_tensorcnt 0
        }
        __syncthreads();
    } else {
        // Async 16B copies (global_load_async_to_lds_b128, ASYNCcnt).
        constexpr int C16_ROW = KDIM / 8;
        constexpr int ITERS = (MBLK * C16_ROW) / NTHREADS;
#pragma unroll
        for (int i = 0; i < ITERS; ++i) {
            const int c = tid + i * NTHREADS;
            const int row = c / C16_ROW, off = c % C16_ROW;
            const unsigned lds_off = (unsigned)(row * (LDSK * 2) + off * 16);
            async_ld_b128(lds_off, Amat + (size_t)(m0 + row) * KDIM + off * 8);
        }
        wait_async0();                      // s_wait_asynccnt 0
        __syncthreads();
    }

    const int wave = tid >> 5, L = tid & 31;
    const int wm = wave % WM, wn = wave / WM;
    const int half = L >> 4;                 // lane half selects K sub-block
    const int lrow = 16 * wm + (L & 15);     // A row inside LDS tile

    const v8f vzero = {0.f, 0.f, 0.f, 0.f, 0.f, 0.f, 0.f, 0.f};
    v8f acc[8];
#pragma unroll
    for (int t = 0; t < 8; ++t) acc[t] = vzero;

    for (int kt = 0; kt < KT; ++kt) {
        // A fragment in contiguous k order: e=0..15 -> k = half*16+e
        //                                   (matches packed B fragments)
        Frag16 af;
        const char* abase = (const char*)lds + (size_t)lrow * (LDSK * 2)
                          + kt * 64 + half * 32;
        af.u[0] = *(const uint4*)(abase);
        af.u[1] = *(const uint4*)(abase + 16);
#pragma unroll
        for (int nt = 0; nt < 8; ++nt) {
            const int tileIdx = kt * NT + (wn * 8 + nt);
            Frag16 bfr;
            const uint4* bp = (const uint4*)(Wfrag + ((size_t)tileIdx * 32 + L) * 16);
            bfr.u[0] = bp[0]; bfr.u[1] = bp[1];
            acc[nt] = __builtin_amdgcn_wmma_f32_16x16x32_bf16(
                false, af.v, false, bfr.v, (short)0, acc[nt], false, false);
        }
        if (kt + 1 < KT)   // pull next k-step's weight fragments toward L2/L0
            __builtin_prefetch(Wfrag + ((size_t)((kt + 1) * NT + wn * 8) * 32) * 16, 0, 0);
    }

    // Epilogue. C/D layout: VGPR r -> row (L>>4)*8+r, col L&15.
    float dtv[8];
    if (SCALE) {
#pragma unroll
        for (int r = 0; r < 8; ++r)
            dtv[r] = dtm[m0 + 16 * wm + half * 8 + r];
    }
#pragma unroll
    for (int nt = 0; nt < 8; ++nt) {
        const int col = 128 * wn + nt * 16 + (L & 15);
#pragma unroll
        for (int r = 0; r < 8; ++r) {
            const int row = m0 + 16 * wm + half * 8 + r;
            float v = acc[nt][r];
            if (SCALE) v *= dtv[r];
            Out[(size_t)row * NTOT + col] = v;
        }
    }
}

// ---------------------------------------------------------------------------
// Scan phase 1: per (b,g,n,chunk) compute chunk aggregate
//   P = prod A_t (complex),  hloc = local recurrence end (h_init = 0).
// A_t = exp(dt_mag*negLogA) * exp(i * dt_phase * A_phase)
// ---------------------------------------------------------------------------
__global__ void scan_phase1(const float* __restrict__ Bx, const float* __restrict__ dtm,
                            const float* __restrict__ dtp, const float* __restrict__ lam,
                            const float* __restrict__ aph, float4* __restrict__ agg) {
    const int q = blockIdx.x * blockDim.x + threadIdx.x;
    const int chunk = q >> 11, r = q & (NSCAN - 1);
    const int b = r >> 9, g = (r >> 7) & 3, n = r & 127;
    const float nla = -log1pf(expf(lam[g * Nn + n]));
    const float ph = aph[g * Nn + n];
    float hr = 0.f, hi = 0.f, Pr = 1.f, Pi = 0.f;
    const int s0 = chunk * LC;
    for (int s = s0; s < s0 + LC; ++s) {
        const int m = (b * Ss + s) * Gg + g;
        const float dm = dtm[m], dp = dtp[m];
        const float amag = __expf(dm * nla);
        float sn, cs; __sincosf(dp * ph, &sn, &cs);
        const float ar = amag * cs, ai = amag * sn;
        const float br = Bx[(size_t)m * N1 + n];
        const float bi = Bx[(size_t)m * N1 + Nn + n];
        const float nhr = ar * hr - ai * hi + br;
        const float nhi = ar * hi + ai * hr + bi;
        hr = nhr; hi = nhi;
        const float nPr = ar * Pr - ai * Pi;
        const float nPi = ar * Pi + ai * Pr;
        Pr = nPr; Pi = nPi;
    }
    agg[(size_t)chunk * NSCAN + r] = make_float4(Pr, Pi, hr, hi);
}

// Scan phase 2: exclusive scan over the 32 chunk aggregates per recurrence.
__global__ void scan_phase2(const float4* __restrict__ agg, float2* __restrict__ hcar) {
    const int q = blockIdx.x * blockDim.x + threadIdx.x;  // 0..2047
    float cr = 0.f, ci = 0.f;
    for (int c = 0; c < NC; ++c) {
        hcar[(size_t)c * NSCAN + q] = make_float2(cr, ci);
        const float4 a = agg[(size_t)c * NSCAN + q];
        const float ncr = a.x * cr - a.y * ci + a.z;
        const float nci = a.x * ci + a.y * cr + a.w;
        cr = ncr; ci = nci;
    }
}

// Scan phase 3: replay each chunk with its carry, apply norm clip, emit bf16 h.
__global__ void scan_phase3(const float* __restrict__ Bx, const float* __restrict__ dtm,
                            const float* __restrict__ dtp, const float* __restrict__ lam,
                            const float* __restrict__ aph, const float2* __restrict__ hcar,
                            bf16* __restrict__ hbuf) {
    const int q = blockIdx.x * blockDim.x + threadIdx.x;
    const int chunk = q >> 11, r = q & (NSCAN - 1);
    const int b = r >> 9, g = (r >> 7) & 3, n = r & 127;
    const float nla = -log1pf(expf(lam[g * Nn + n]));
    const float ph = aph[g * Nn + n];
    const float2 c0 = hcar[(size_t)chunk * NSCAN + r];
    float hr = c0.x, hi = c0.y;
    const int s0 = chunk * LC;
    for (int s = s0; s < s0 + LC; ++s) {
        const int m = (b * Ss + s) * Gg + g;
        const float dm = dtm[m], dp = dtp[m];
        const float amag = __expf(dm * nla);
        float sn, cs; __sincosf(dp * ph, &sn, &cs);
        const float ar = amag * cs, ai = amag * sn;
        const float br = Bx[(size_t)m * N1 + n];
        const float bi = Bx[(size_t)m * N1 + Nn + n];
        const float nhr = ar * hr - ai * hi + br;
        const float nhi = ar * hi + ai * hr + bi;
        hr = nhr; hi = nhi;
        const float nrm = sqrtf(hr * hr + hi * hi + 1e-8f);
        const float scl = fminf(nrm, 100.f) / nrm;   // clip on output only
        hbuf[(size_t)m * K2 + n]      = (bf16)(hr * scl);
        hbuf[(size_t)m * K2 + Nn + n] = (bf16)(hi * scl);
    }
}

// ---------------------------------------------------------------------------
extern "C" void kernel_launch(void* const* d_in, const int* in_sizes, int n_in,
                              void* d_out, int out_size, void* d_ws, size_t ws_size,
                              hipStream_t stream) {
    const float* xr  = (const float*)d_in[0];
    const float* xi  = (const float*)d_in[1];
    const float* lam = (const float*)d_in[2];
    const float* aph = (const float*)d_in[3];
    const float* bwr = (const float*)d_in[4];
    const float* bwi = (const float*)d_in[5];
    const float* cwr = (const float*)d_in[6];
    const float* cwi = (const float*)d_in[7];
    const float* dtw = (const float*)d_in[8];
    const float* dtb = (const float*)d_in[9];

    char* ws = (char*)d_ws;
    size_t off = 0;
    auto alloc = [&](size_t bytes) -> void* {
        void* p = ws + off;
        off = (off + bytes + 255) & ~(size_t)255;
        return p;
    };
    bf16*   xcat = (bf16*)  alloc((size_t)M * K1 * sizeof(bf16));   // 64 MB
    bf16*   wbf  = (bf16*)  alloc((size_t)K1 * N1 * sizeof(bf16));  // 256 KB
    bf16*   wcf  = (bf16*)  alloc((size_t)K2 * N2 * sizeof(bf16));  // 256 KB
    float*  dtm  = (float*) alloc((size_t)M * sizeof(float));       // 256 KB
    float*  dtp  = (float*) alloc((size_t)M * sizeof(float));       // 256 KB
    float*  Bx   = (float*) alloc((size_t)M * N1 * sizeof(float));  // 64 MB
    bf16*   hbuf = (bf16*)  alloc((size_t)M * K2 * sizeof(bf16));   // 32 MB
    float4* agg  = (float4*)alloc((size_t)NC * NSCAN * sizeof(float4)); // 1 MB
    float2* hcar = (float2*)alloc((size_t)NC * NSCAN * sizeof(float2)); // 512 KB
    (void)ws_size; (void)in_sizes; (void)n_in; (void)out_size;

    // 0) pack weights into WMMA fragment order
    prep_weights<<<512, 32, 0, stream>>>(bwr, bwi, cwr, cwi, wbf, wcf);
    // 1) bf16 conversion of x + dt head
    prep_x_dt<<<M / 8, 256, 0, stream>>>(xr, xi, dtw, dtb, xcat, dtm, dtp);
    // 2) input projection: Bx[M,256] = xcat @ W_B, scaled by dt_mag (TDM stage)
    gemm_bf16_wmma<K1, N1, 32, 2, 2, true, true>
        <<<M / 32, 128, 0, stream>>>(xcat, wbf, dtm, Bx);
    // 3) chunked complex scan (3 phases)
    scan_phase1<<<(NC * NSCAN) / 256, 256, 0, stream>>>(Bx, dtm, dtp, lam, aph, agg);
    scan_phase2<<<NSCAN / 256, 256, 0, stream>>>(agg, hcar);
    scan_phase3<<<(NC * NSCAN) / 256, 256, 0, stream>>>(Bx, dtm, dtp, lam, aph, hcar, hbuf);
    // 4) output projection: y[M,512] = hbuf @ W_C  (async-LDS stage)
    gemm_bf16_wmma<K2, N2, 32, 2, 4, false, false>
        <<<M / 32, 256, 0, stream>>>(hbuf, wcf, nullptr, (float*)d_out);
}
